// MultiHeadAttention_68504728371772
// MI455X (gfx1250) — matrix operand order repaired and checked by reading, not compiled, not run on hardware
//
#include <hip/hip_runtime.h>
#include <hip/hip_bf16.h>

typedef __attribute__((ext_vector_type(16))) _Float16 v16h;
typedef __attribute__((ext_vector_type(8)))  _Float16 v8h;
typedef __attribute__((ext_vector_type(4)))  _Float16 v4h;
typedef __attribute__((ext_vector_type(8)))  float    v8f;

#define WMMA_F16(a, b, c) \
    __builtin_amdgcn_wmma_f32_16x16x32_f16(false, (a), false, (b), (short)0, (c), false, false)

static constexpr int B_SZ   = 2;
static constexpr int L_SEQ  = 4096;
static constexpr int D_MODEL= 256;
static constexpr int N_HEADS= 8;
static constexpr int D_K    = 32;
static constexpr int M_ROWS = B_SZ * L_SEQ;                 // 8192
static constexpr int MAT_ELEMS  = M_ROWS * D_MODEL;         // 2,097,152 per Q/K/V matrix (f16)
static constexpr int HEAD_ELEMS = L_SEQ * D_K;              // 131,072 per (b,h)

// ---------------------------------------------------------------------------
// Kernel 1: fused Q/K/V projection.  y = x @ W^T + b, stored f16 head-major.
// grid = (M/16, 3), block = 256 (8 waves).  Wave w owns output cols [32w,32w+32).
// ---------------------------------------------------------------------------
__global__ __launch_bounds__(256) void qkv_proj_kernel(
    const float* __restrict__ x,
    const float* __restrict__ Wq, const float* __restrict__ bq,
    const float* __restrict__ Wk, const float* __restrict__ bk,
    const float* __restrict__ Wv, const float* __restrict__ bv,
    _Float16* __restrict__ qkv_ws)
{
    const int lane = threadIdx.x & 31;
    const int wave = threadIdx.x >> 5;
    const int half = lane >> 4;
    const int mr   = lane & 15;
    const int M0   = blockIdx.x * 16;

    const float* W;  const float* bias;
    if (blockIdx.y == 0)      { W = Wq; bias = bq; }
    else if (blockIdx.y == 1) { W = Wk; bias = bk; }
    else                      { W = Wv; bias = bv; }

    const int n0 = wave * 32 + mr;     // this lane's column in N-subtile 0
    const int n1 = n0 + 16;            // ... in N-subtile 1
    const float* xrow  = x + (size_t)(M0 + mr) * D_MODEL;
    const float* wrow0 = W + (size_t)n0 * D_MODEL;
    const float* wrow1 = W + (size_t)n1 * D_MODEL;

    v8f c0 = {}; v8f c1 = {};

    #pragma unroll
    for (int kk = 0; kk < D_MODEL; kk += 32) {
        // A fragment: a[i] <-> (K = kk + 16*half + i)
        v16h a;
        #pragma unroll
        for (int i = 0; i < 16; ++i) a[i] = (_Float16)xrow[kk + 16 * half + i];
        // B fragments: b[i] <-> (K = kk + 16*half + i, N = n)
        v16h b0, b1;
        #pragma unroll
        for (int i = 0; i < 16; ++i) b0[i] = (_Float16)wrow0[kk + 16 * half + i];
        #pragma unroll
        for (int i = 0; i < 16; ++i) b1[i] = (_Float16)wrow1[kk + 16 * half + i];
        c0 = WMMA_F16(a, b0, c0);
        c1 = WMMA_F16(a, b1, c1);
    }

    const float bv0 = bias[n0], bv1 = bias[n1];
    const int h0 = n0 >> 5, d0 = n0 & 31;
    const int h1 = n1 >> 5, d1 = n1 & 31;
    _Float16* outp = qkv_ws + (size_t)blockIdx.y * MAT_ELEMS;
    #pragma unroll
    for (int r = 0; r < 8; ++r) {
        const int mg = M0 + r + 8 * half;          // global row = b*L + l
        const int bb = mg >> 12, ll = mg & 4095;
        outp[((size_t)(bb * N_HEADS + h0) * L_SEQ + ll) * D_K + d0] = (_Float16)(c0[r] + bv0);
        outp[((size_t)(bb * N_HEADS + h1) * L_SEQ + ll) * D_K + d1] = (_Float16)(c1[r] + bv1);
    }
}

// ---------------------------------------------------------------------------
// Kernel 2: flash attention.  grid = (L/128, B*H), block = 256 (8 waves).
// Each wave owns 16 query rows; K/V tiles streamed into LDS with gfx1250
// async-to-LDS loads (ASYNCcnt), softmax co-executes with XDL WMMAs.
// ---------------------------------------------------------------------------
__global__ __launch_bounds__(256) void flash_attn_kernel(
    const _Float16* __restrict__ qkv_ws,
    _Float16* __restrict__ attn_ws)
{
    __shared__ _Float16 Kl[32][40];       // padded stride 40 halves (80B) -> bank spread
    __shared__ _Float16 Vl[32][40];
    __shared__ _Float16 Pl[8][16][40];    // per-wave C-layout -> A-layout staging

    const int lane = threadIdx.x & 31;
    const int wave = threadIdx.x >> 5;
    const int half = lane >> 4;
    const int mr   = lane & 15;
    const int bh   = blockIdx.y;

    const _Float16* Qh = qkv_ws                 + (size_t)bh * HEAD_ELEMS;
    const _Float16* Kh = qkv_ws + MAT_ELEMS     + (size_t)bh * HEAD_ELEMS;
    const _Float16* Vh = qkv_ws + 2 * MAT_ELEMS + (size_t)bh * HEAD_ELEMS;

    const int q0 = blockIdx.x * 128 + wave * 16;

    // Q fragment, pre-scaled by 1/sqrt(D_K)
    const float scale = 0.17677669529663687f;
    const _Float16* qrow = Qh + (size_t)(q0 + mr) * D_K;
    v16h qa;
    #pragma unroll
    for (int i = 0; i < 16; ++i) qa[i] = (_Float16)((float)qrow[16 * half + i] * scale);

    v8f o0 = {}; v8f o1 = {};
    float mrun[8], lrun[8];
    #pragma unroll
    for (int r = 0; r < 8; ++r) { mrun[r] = -1e30f; lrun[r] = 0.f; }

    const int ldr = threadIdx.x >> 3;        // 0..31 : kv row for cooperative load
    const int ldc = (threadIdx.x & 7) * 4;   // 0..28 : 4-half column chunk

    // LDS byte offsets for this thread's async-load destinations
    // (LDS aperture truncates flat addr to addr[31:0] => low 32 bits = DS offset)
    const unsigned k_lds = (unsigned)(uintptr_t)(&Kl[ldr][ldc]);
    const unsigned v_lds = (unsigned)(uintptr_t)(&Vl[ldr][ldc]);

    for (int kv0 = 0; kv0 < L_SEQ; kv0 += 32) {
        __syncthreads();
        // gfx1250 async memory->LDS (8B/lane x 256 lanes = 2KB per matrix tile)
        {
            const unsigned long long kga =
                (unsigned long long)(uintptr_t)(Kh + (size_t)(kv0 + ldr) * D_K + ldc);
            const unsigned long long vga =
                (unsigned long long)(uintptr_t)(Vh + (size_t)(kv0 + ldr) * D_K + ldc);
            asm volatile("global_load_async_to_lds_b64 %0, %1, off"
                         :: "v"(k_lds), "v"(kga) : "memory");
            asm volatile("global_load_async_to_lds_b64 %0, %1, off"
                         :: "v"(v_lds), "v"(vga) : "memory");
        }
        if (kv0 + 32 < L_SEQ) {  // -> global_prefetch_b8 for next tile
            __builtin_prefetch(Kh + (size_t)(kv0 + 32 + ldr) * D_K + ldc, 0, 1);
            __builtin_prefetch(Vh + (size_t)(kv0 + 32 + ldr) * D_K + ldc, 0, 1);
        }
        asm volatile("s_wait_asynccnt 0x0" ::: "memory");
        __syncthreads();

        // S = Q K^T : B[d][kv] = K[kv][d];  b[i] <-> (d = 16*half+i, kv = N0+mr)
        v16h kb0, kb1;
        #pragma unroll
        for (int i = 0; i < 16; ++i) kb0[i] = Kl[mr][16 * half + i];
        #pragma unroll
        for (int i = 0; i < 16; ++i) kb1[i] = Kl[16 + mr][16 * half + i];
        const v8f zero = {};
        v8f s0 = WMMA_F16(qa, kb0, zero);
        v8f s1 = WMMA_F16(qa, kb1, zero);

        // online softmax: row stats live across lanes 0-15 within each half
        float sm[8];
        #pragma unroll
        for (int r = 0; r < 8; ++r) sm[r] = fmaxf(s0[r], s1[r]);
        #pragma unroll
        for (int off = 8; off >= 1; off >>= 1)
            #pragma unroll
            for (int r = 0; r < 8; ++r) sm[r] = fmaxf(sm[r], __shfl_xor(sm[r], off, 32));

        float alpha[8], rs[8];
        v8f p0, p1;
        #pragma unroll
        for (int r = 0; r < 8; ++r) {
            const float mn = fmaxf(mrun[r], sm[r]);
            alpha[r] = __expf(mrun[r] - mn);
            mrun[r]  = mn;
            p0[r] = __expf(s0[r] - mn);
            p1[r] = __expf(s1[r] - mn);
            rs[r] = p0[r] + p1[r];
        }
        #pragma unroll
        for (int off = 8; off >= 1; off >>= 1)
            #pragma unroll
            for (int r = 0; r < 8; ++r) rs[r] += __shfl_xor(rs[r], off, 32);
        #pragma unroll
        for (int r = 0; r < 8; ++r) {
            lrun[r] = lrun[r] * alpha[r] + rs[r];
            o0[r] *= alpha[r];
            o1[r] *= alpha[r];
        }

        // P: C-layout -> LDS -> A-layout (intra-wave, DS ops are in-order per wave)
        #pragma unroll
        for (int r = 0; r < 8; ++r) {
            Pl[wave][r + 8 * half][mr]      = (_Float16)p0[r];
            Pl[wave][r + 8 * half][16 + mr] = (_Float16)p1[r];
        }
        v16h pa;
        #pragma unroll
        for (int i = 0; i < 16; ++i) pa[i] = Pl[wave][mr][16 * half + i];

        // O += P V : b[i] <-> (kv = 16*half+i, d = N0+mr)
        v16h vb0, vb1;
        #pragma unroll
        for (int i = 0; i < 16; ++i) vb0[i] = Vl[16 * half + i][mr];
        #pragma unroll
        for (int i = 0; i < 16; ++i) vb1[i] = Vl[16 * half + i][16 + mr];
        o0 = WMMA_F16(pa, vb0, o0);
        o1 = WMMA_F16(pa, vb1, o1);
    }

    // normalize + store f16 in [b][l][h*32+d] so the O-projection reads rows
    const int b = bh >> 3, h = bh & 7;
    #pragma unroll
    for (int r = 0; r < 8; ++r) {
        const float inv = (lrun[r] > 0.f) ? (1.f / lrun[r]) : 0.f;
        const int lp = q0 + r + 8 * half;
        const size_t base = ((size_t)(b * L_SEQ + lp)) * D_MODEL + h * D_K;
        attn_ws[base + mr]      = (_Float16)(o0[r] * inv);
        attn_ws[base + 16 + mr] = (_Float16)(o1[r] * inv);
    }
}

// ---------------------------------------------------------------------------
// Kernel 3: output projection.  out = attn @ Wo^T + bo, fp32 result.
// grid = M/16, block = 256 (8 waves).
// ---------------------------------------------------------------------------
__global__ __launch_bounds__(256) void out_proj_kernel(
    const _Float16* __restrict__ xin,
    const float* __restrict__ Wo, const float* __restrict__ bo,
    float* __restrict__ out)
{
    const int lane = threadIdx.x & 31;
    const int wave = threadIdx.x >> 5;
    const int half = lane >> 4;
    const int mr   = lane & 15;
    const int M0   = blockIdx.x * 16;

    const int n0 = wave * 32 + mr;
    const int n1 = n0 + 16;
    const _Float16* xrow = xin + (size_t)(M0 + mr) * D_MODEL;
    const float* wrow0 = Wo + (size_t)n0 * D_MODEL;
    const float* wrow1 = Wo + (size_t)n1 * D_MODEL;

    v8f c0 = {}; v8f c1 = {};
    #pragma unroll
    for (int kk = 0; kk < D_MODEL; kk += 32) {
        v16h a;
        #pragma unroll
        for (int i = 0; i < 16; ++i) a[i] = xrow[kk + 16 * half + i];
        v16h b0, b1;
        #pragma unroll
        for (int i = 0; i < 16; ++i) b0[i] = (_Float16)wrow0[kk + 16 * half + i];
        #pragma unroll
        for (int i = 0; i < 16; ++i) b1[i] = (_Float16)wrow1[kk + 16 * half + i];
        c0 = WMMA_F16(a, b0, c0);
        c1 = WMMA_F16(a, b1, c1);
    }

    const float bv0 = bo[n0], bv1 = bo[n1];
    #pragma unroll
    for (int r = 0; r < 8; ++r) {
        const size_t row = (size_t)(M0 + r + 8 * half) * D_MODEL;
        out[row + n0] = c0[r] + bv0;
        out[row + n1] = c1[r] + bv1;
    }
}

// ---------------------------------------------------------------------------
extern "C" void kernel_launch(void* const* d_in, const int* in_sizes, int n_in,
                              void* d_out, int out_size, void* d_ws, size_t ws_size,
                              hipStream_t stream) {
    const float* x  = (const float*)d_in[0];
    const float* Wq = (const float*)d_in[1];
    const float* bq = (const float*)d_in[2];
    const float* Wk = (const float*)d_in[3];
    const float* bk = (const float*)d_in[4];
    const float* Wv = (const float*)d_in[5];
    const float* bv = (const float*)d_in[6];
    const float* Wo = (const float*)d_in[7];
    const float* bo = (const float*)d_in[8];

    _Float16* qkv  = (_Float16*)d_ws;                 // 3 * 2,097,152 f16 = 12.6 MB
    _Float16* attn = qkv + 3 * (size_t)MAT_ELEMS;     // 2,097,152 f16 = 4.2 MB

    dim3 g1(M_ROWS / 16, 3);
    qkv_proj_kernel<<<g1, 256, 0, stream>>>(x, Wq, bq, Wk, bk, Wv, bv, qkv);

    dim3 g2(L_SEQ / 128, B_SZ * N_HEADS);
    flash_attn_kernel<<<g2, 256, 0, stream>>>(qkv, attn);

    out_proj_kernel<<<dim3(M_ROWS / 16), 256, 0, stream>>>(attn, Wo, bo, (float*)d_out);
}
